// VanillaRNN_26268019982886
// MI455X (gfx1250) — matrix operand order repaired and checked by reading, not compiled, not run on hardware
//
#include <hip/hip_runtime.h>
#include <hip/hip_bf16.h>
#include <math.h>

typedef __bf16 bf16_t;
typedef __attribute__((ext_vector_type(16))) __bf16 v16bf;
typedef __attribute__((ext_vector_type(8)))  __bf16 v8bf;
typedef __attribute__((ext_vector_type(8)))  float  v8f;
typedef __attribute__((ext_vector_type(4)))  unsigned int u32x4;
typedef __attribute__((ext_vector_type(8)))  int i32x8;
typedef __attribute__((ext_vector_type(4)))  int i32x4;

#define HID   1024
#define BATCH 512
#define SEQ   512
// LDS row stride: 1024 bf16 = 2048B + TDM pad (2 dwords every 256 dwords)
// => 512 + 4 dwords = 2064 bytes: 16B aligned, lanes land on banks 4*r (conflict-free b128).
#define ROWB  2064
#define MROWS 32     // W rows staged per block (two 16-row M tiles per wave)

// Branch-free tanh: (e^2x - 1)/(e^2x + 1) via v_exp_f32 + v_rcp_f32.
// Clamped to +-9 so e^2x stays finite (no inf/inf); error << bf16 ulp.
__device__ __forceinline__ float fast_tanh(float v)
{
    v = fminf(fmaxf(v, -9.0f), 9.0f);
    float a = __builtin_amdgcn_exp2f(v * 2.8853900817779268f);  // e^(2v)
    return (a - 1.0f) * __builtin_amdgcn_rcpf(a + 1.0f);
}

// ---------------- step kernel: hT_out = tanh(W_hh @ h + W_hx x_t^T + b_h), transposed layout
__global__ __launch_bounds__(256) void rnn_step_kernel(
    const bf16_t* __restrict__ Wbf,   // [1024,1024] bf16 row-major (m,k)
    const bf16_t* __restrict__ hin,   // hT: [512,1024] bf16 (b,k)
    bf16_t* __restrict__ hout,        // hT_new: [512,1024]
    const float* __restrict__ x,      // [512,512] (b,t)
    const float* __restrict__ Whx,    // [1024]
    const float* __restrict__ bh,     // [512]
    int t)
{
    __shared__ char smem[MROWS * ROWB];

    const int wave = threadIdx.x >> 5;
    const int lane = threadIdx.x & 31;
    const int half = lane >> 4;      // 0: lanes 0-15, 1: lanes 16-31
    const int lr   = lane & 15;
    const int m0   = blockIdx.y * MROWS;              // 32 row strips of 32
    const int n0   = (blockIdx.x * 8 + wave) * 16;    // 4 blocks x 8 waves = 32 col tiles

    // Wave 0 DMAs the 32x1024 bf16 W-strip for rows [m0, m0+32) into LDS via TDM.
    if (wave == 0) {
        unsigned long long ga = (unsigned long long)(uintptr_t)(Wbf + (size_t)m0 * HID);
        unsigned int ldsoff = (unsigned int)(unsigned long long)(uintptr_t)(&smem[0]);
        u32x4 g0;
        g0[0] = 1u;                                   // count=1, user descriptor
        g0[1] = ldsoff;                               // lds_addr (bytes)
        g0[2] = (unsigned int)ga;                     // global_addr[31:0]
        g0[3] = (unsigned int)((ga >> 32) & 0x1FFFFFFu) | (2u << 30); // addr[56:32] | type=2
        i32x8 g1;
        g1[0] = (1 << 16)      // data_size = 2 bytes
              | (1 << 20)      // pad_enable
              | (7 << 22)      // pad_interval: every 256 dwords
              | (1 << 25);     // pad_amount: 2 dwords
        g1[1] = (int)(((unsigned)HID & 0xFFFFu) << 16);  // tensor_dim0[15:0]=1024
        g1[2] = (int)(((unsigned)HID & 0xFFFFu) << 16);  // tensor_dim1[15:0]=1024
        g1[3] = (int)(((unsigned)HID & 0xFFFFu) << 16);  // tile_dim0=1024
        g1[4] = MROWS;                                   // tile_dim1=32 rows
        g1[5] = HID;                                     // tensor_dim0_stride=1024
        g1[6] = 0;
        g1[7] = 0;
        i32x4 gz4 = {0, 0, 0, 0};
        i32x8 gz8 = {0, 0, 0, 0, 0, 0, 0, 0};
        __builtin_amdgcn_tensor_load_to_lds(g0, g1, gz4, gz4, gz8, 0);
        __builtin_amdgcn_s_wait_tensorcnt(0);
    }
    __syncthreads();

    // B-fragment source: one hT row per lane (column n of the output tile).
    const int n = n0 + lr;
    const bf16_t* __restrict__ hrow = hin + (size_t)n * HID;
    const char* __restrict__ arow0 = smem + lr * ROWB;          // rows m0..m0+15
    const char* __restrict__ arow1 = smem + (16 + lr) * ROWB;   // rows m0+16..m0+31

    v8f acc0 = {0.f, 0.f, 0.f, 0.f, 0.f, 0.f, 0.f, 0.f};
    v8f acc1 = {0.f, 0.f, 0.f, 0.f, 0.f, 0.f, 0.f, 0.f};

#pragma unroll 2
    for (int kc = 0; kc < HID; kc += 32) {
        // B fragment 32x16 bf16: lane(lr,half) holds col n0+lr, K = kc+16*half..+15
        // (contiguous 32B) -- loaded ONCE, reused by both M tiles.
        v16bf Bf = *(const v16bf*)(hrow + kc + 16 * half);
        // A fragments 16x32 bf16: lane(lr,half) holds row m,
        // K = kc+16*half..+7 (v0..v3) and kc+16*half+8..+15 (v4..v7).
        union { v16bf v; v8bf h[2]; } A0, A1;
        A0.h[0] = *(const v8bf*)(arow0 + (kc + 16 * half) * 2);
        A0.h[1] = *(const v8bf*)(arow0 + (kc + 16 * half + 8) * 2);
        A1.h[0] = *(const v8bf*)(arow1 + (kc + 16 * half) * 2);
        A1.h[1] = *(const v8bf*)(arow1 + (kc + 16 * half + 8) * 2);
        acc0 = __builtin_amdgcn_wmma_f32_16x16x32_bf16(
            false, A0.v, false, Bf, (short)0, acc0, false, false);
        acc1 = __builtin_amdgcn_wmma_f32_16x16x32_bf16(
            false, A1.v, false, Bf, (short)0, acc1, false, false);
    }

    // Epilogue: D layout => lane holds column n, VGPR j => row m + 8*half + j.
    const float xt = x[(size_t)n * SEQ + t];
    const float bb = bh[n];
    const float* __restrict__ wrow0 = Whx + m0 + 8 * half;
    const float* __restrict__ wrow1 = Whx + m0 + 16 + 8 * half;
    bf16_t* __restrict__ orow = hout + (size_t)n * HID;
    v8bf o0, o1;
#pragma unroll
    for (int j = 0; j < 8; ++j) {
        o0[j] = (bf16_t)fast_tanh(acc0[j] + wrow0[j] * xt + bb);
        o1[j] = (bf16_t)fast_tanh(acc1[j] + wrow1[j] * xt + bb);
    }
    // 8 consecutive m per lane -> packed 16B stores into hT_new row n.
    *(v8bf*)(orow + m0 + 8 * half)      = o0;
    *(v8bf*)(orow + m0 + 16 + 8 * half) = o1;
}

// ---------------- W_hh f32 -> bf16
__global__ __launch_bounds__(256) void cvt_bf16_kernel(
    const float* __restrict__ in, bf16_t* __restrict__ out, int nel)
{
    int i = blockIdx.x * blockDim.x + threadIdx.x;
    if (i < nel) out[i] = (bf16_t)in[i];
}

// ---------------- zero h0 (1 MB), 16B per thread
__global__ __launch_bounds__(256) void zero_kernel(uint4* __restrict__ p)
{
    int i = blockIdx.x * blockDim.x + threadIdx.x;
    uint4 z = {0u, 0u, 0u, 0u};
    p[i] = z;
}

// ---------------- out[b, c] = W_ph[c,:] . h[:, b] + b_p[b]   (tiny: 10 classes)
__global__ __launch_bounds__(256) void proj_kernel(
    const bf16_t* __restrict__ hT, const float* __restrict__ Wph,
    const float* __restrict__ bp, float* __restrict__ out)
{
    int idx = blockIdx.x * blockDim.x + threadIdx.x;
    if (idx >= BATCH * 10) return;
    int n = idx / 10, cls = idx % 10;
    const bf16_t* __restrict__ hr = hT + (size_t)n * HID;
    const float* __restrict__ wr = Wph + (size_t)cls * HID;
    float s = bp[n];
#pragma unroll 8
    for (int k = 0; k < HID; ++k) s += wr[k] * (float)hr[k];
    out[(size_t)n * 10 + cls] = s;
}

extern "C" void kernel_launch(void* const* d_in, const int* in_sizes, int n_in,
                              void* d_out, int out_size, void* d_ws, size_t ws_size,
                              hipStream_t stream)
{
    const float* x    = (const float*)d_in[0];  // [512,512]
    const float* Whx  = (const float*)d_in[1];  // [1024,1]
    const float* Whh  = (const float*)d_in[2];  // [1024,1024]
    const float* Wph  = (const float*)d_in[3];  // [10,1024]
    const float* bh   = (const float*)d_in[4];  // [512]
    const float* bp   = (const float*)d_in[5];  // [512]
    (void)in_sizes; (void)n_in; (void)out_size; (void)ws_size;

    char* ws = (char*)d_ws;
    bf16_t* Wbf = (bf16_t*)ws;                          // 2 MB
    bf16_t* h0  = (bf16_t*)(ws + (2ull << 20));         // 1 MB
    bf16_t* h1  = (bf16_t*)(ws + (3ull << 20));         // 1 MB

    cvt_bf16_kernel<<<(HID * HID) / 256, 256, 0, stream>>>(Whh, Wbf, HID * HID);
    zero_kernel<<<(BATCH * HID * 2 / 16) / 256, 256, 0, stream>>>((uint4*)h0);

    for (int t = 0; t < SEQ; ++t) {
        bf16_t* hin  = (t & 1) ? h1 : h0;
        bf16_t* hout = (t & 1) ? h0 : h1;
        rnn_step_kernel<<<dim3(4, HID / MROWS), 256, 0, stream>>>(Wbf, hin, hout, x, Whx, bh, t);
    }
    // SEQ=512 even: final h lands in h0
    proj_kernel<<<(BATCH * 10 + 255) / 256, 256, 0, stream>>>(h0, Wph, bp, (float*)d_out);
}
